// SimpleMoeBlock_13030930776882
// MI455X (gfx1250) — compile-verified, matrix operand order checked by reading, not run on hardware
//
#include <hip/hip_runtime.h>
#include <hip/hip_bf16.h>

// MoE: B=2,S=2048 -> N=4096 tokens, H=2048, E=8, K=2(top-k), I=5632
#define NTOK 4096
#define HDIM 2048
#define IDIM 5632
#define NEXP 8

typedef __attribute__((ext_vector_type(16))) __bf16 v16bf;
typedef __attribute__((ext_vector_type(8)))  __bf16 v8bf;
typedef __attribute__((ext_vector_type(8)))  float  v8f;

#define LDST 40   // LDS row stride (bf16 elems): 32 payload + 8 pad -> 80B rows, conflict-free
#define EPST 72   // epilogue staging row stride (bf16): 64 payload + 8 pad (144B, 16B-aligned)

// 16x32 bf16 WMMA fragment from an LDS tile (row-major, K-contiguous, stride LDST).
// ISA layout: lanes 0-15: row=lane, K{0..7,16..23}; lanes 16-31: row=lane-16, K{8..15,24..31}.
__device__ __forceinline__ v16bf frag_ld(const __bf16* base, int lane) {
  const int half = lane >> 4;
  const int r    = lane & 15;
  const __bf16* p = base + r * LDST + half * 8;
  v8bf lo = *(const v8bf*)(p);
  v8bf hi = *(const v8bf*)(p + 16);
  return __builtin_shufflevector(lo, hi, 0,1,2,3,4,5,6,7,8,9,10,11,12,13,14,15);
}

__device__ __forceinline__ v8bf cvt8(float4 a, float4 b) {
  v8bf v;
  v[0]=(__bf16)a.x; v[1]=(__bf16)a.y; v[2]=(__bf16)a.z; v[3]=(__bf16)a.w;
  v[4]=(__bf16)b.x; v[5]=(__bf16)b.y; v[6]=(__bf16)b.z; v[7]=(__bf16)b.w;
  return v;
}

__device__ __forceinline__ v8f wmma_bf16(v16bf a, v16bf b, v8f c) {
  return __builtin_amdgcn_wmma_f32_16x16x32_bf16(false, a, false, b, (short)0, c,
                                                 false, false);
}

// Async HBM->LDS 16B copy (no VGPR staging), tracked by ASYNCcnt.
__device__ __forceinline__ void async_ld_b128(unsigned lds_off,
                                              const void* gaddr) {
  asm volatile("global_load_async_to_lds_b128 %0, %1, off"
               :: "v"(lds_off), "v"((unsigned long long)gaddr) : "memory");
}
__device__ __forceinline__ void wait_asynccnt0() {
  asm volatile("s_wait_asynccnt 0" ::: "memory");
}

// ---------------------------------------------------------------- zero + cnt
__global__ __launch_bounds__(256)
void zero_out_k(float4* __restrict__ out, int n4, int* __restrict__ cnt) {
  if (blockIdx.x == 0 && threadIdx.x < NEXP) cnt[threadIdx.x] = 0;
  float4 z; z.x = z.y = z.z = z.w = 0.f;
  for (int i = blockIdx.x * blockDim.x + threadIdx.x; i < n4;
       i += gridDim.x * blockDim.x)
    out[i] = z;
}

// ---------------------------------------------------------------- router
__global__ __launch_bounds__(256)
void router_k(const float* __restrict__ x, const float* __restrict__ rw,
              int* __restrict__ cnt, int* __restrict__ pairTok,
              float* __restrict__ pairW) {
  const int lane = threadIdx.x & 31;
  const int n = blockIdx.x * 8 + (threadIdx.x >> 5);
  if (n >= NTOK) return;

  float acc[NEXP];
#pragma unroll
  for (int e = 0; e < NEXP; ++e) acc[e] = 0.f;
  const float* xp = x + (long)n * HDIM;
  for (int h = lane; h < HDIM; h += 32) {
    float xv = xp[h];
#pragma unroll
    for (int e = 0; e < NEXP; ++e) acc[e] += xv * rw[e * HDIM + h];
  }
#pragma unroll
  for (int e = 0; e < NEXP; ++e) {
#pragma unroll
    for (int off = 16; off; off >>= 1) acc[e] += __shfl_xor(acc[e], off, 32);
  }
  if (lane == 0) {
    float m = acc[0];
#pragma unroll
    for (int e = 1; e < NEXP; ++e) m = fmaxf(m, acc[e]);
    float p[NEXP], s = 0.f;
#pragma unroll
    for (int e = 0; e < NEXP; ++e) { p[e] = __expf(acc[e] - m); s += p[e]; }
    const float inv = 1.f / s;
    int i0 = 0; float b0 = acc[0];
#pragma unroll
    for (int e = 1; e < NEXP; ++e)
      if (acc[e] > b0) { b0 = acc[e]; i0 = e; }
    int i1 = (i0 == 0) ? 1 : 0; float b1 = acc[i1];
#pragma unroll
    for (int e = 0; e < NEXP; ++e)
      if (e != i0 && acc[e] > b1) { b1 = acc[e]; i1 = e; }

    int pos = atomicAdd(&cnt[i0], 1);
    pairTok[i0 * NTOK + pos] = (n << 1);
    pairW[i0 * NTOK + pos] = p[i0] * inv;
    pos = atomicAdd(&cnt[i1], 1);
    pairTok[i1 * NTOK + pos] = (n << 1) | 1;
    pairW[i1 * NTOK + pos] = p[i1] * inv;
  }
}

// ---------------------------------------------------------------- gate+up
// Block tile 128(M) x 64(I) for BOTH gate and up. 8 waves; wave (wr,wcol) owns
// rows 32*wr..+31 (2 M-tiles) and cols 32*wcol..+31 (2 N-tiles per matrix):
// 8 WMMA per K-step vs 12 ds_load_b128 (1.5 ds/wmma).
__global__ __launch_bounds__(256)
void moe_gateup_k(const float* __restrict__ x, const float* __restrict__ gate_w,
                  const float* __restrict__ up_w, const int* __restrict__ cnt,
                  const int* __restrict__ pairTok, __bf16* __restrict__ act) {
  const int e = blockIdx.z;
  const int c = cnt[e];
  const int mBase = blockIdx.y * 128;
  if (mBase >= c) return;
  const int i0 = blockIdx.x * 64;

  __shared__ __bf16 lA[2][128 * LDST];
  __shared__ __bf16 lG[2][64 * LDST];
  __shared__ __bf16 lU[2][64 * LDST];
  __shared__ __bf16 sEpi[128 * EPST];
  __shared__ int sRow[128];

  const int tid = threadIdx.x;
  if (tid < 128) {
    int idx = mBase + tid;
    if (idx > c - 1) idx = c - 1;          // clamp partial tile (stores guarded)
    sRow[tid] = pairTok[e * NTOK + idx];
  }
  __syncthreads();

  // staging maps: A 128x32 fp32 (16 floats/thread), B 64x32 fp32 x2 (8 each)
  const int ra = tid >> 1;                 // 0..127
  const int ca = (tid & 1) * 16;
  const int rb = tid >> 2;                 // 0..63
  const int cb = (tid & 3) * 8;
  const long tok = (long)(sRow[ra] >> 1);
  const float* xp = x + tok * HDIM + ca;
  const float* gp = gate_w + (long)e * IDIM * HDIM + (long)(i0 + rb) * HDIM + cb;
  const float* up = up_w   + (long)e * IDIM * HDIM + (long)(i0 + rb) * HDIM + cb;

  const int lane = tid & 31;
  const int wave = tid >> 5;
  const int wr = wave & 3;                 // rows 32*wr..+31
  const int wcol = wave >> 2;              // cols 32*wcol..+31

  v8f aG[2][2] = {{{}, {}}, {{}, {}}};
  v8f aU[2][2] = {{{}, {}}, {{}, {}}};

  // stage K-step 0 into buffer 0
  {
    float4 a0 = *(const float4*)(xp),     a1 = *(const float4*)(xp + 4);
    float4 a2 = *(const float4*)(xp + 8), a3 = *(const float4*)(xp + 12);
    float4 g0 = *(const float4*)(gp),     g1 = *(const float4*)(gp + 4);
    float4 u0 = *(const float4*)(up),     u1 = *(const float4*)(up + 4);
    *(v8bf*)&lA[0][ra * LDST + ca]     = cvt8(a0, a1);
    *(v8bf*)&lA[0][ra * LDST + ca + 8] = cvt8(a2, a3);
    *(v8bf*)&lG[0][rb * LDST + cb]     = cvt8(g0, g1);
    *(v8bf*)&lU[0][rb * LDST + cb]     = cvt8(u0, u1);
  }

  const int KS = HDIM / 32;
  for (int kt = 0; kt < KS; ++kt) {
    const int cur = kt & 1;
    float4 a0, a1, a2, a3, g0, g1, u0, u1;
    const bool more = (kt + 1) < KS;
    if (more) {
      const int kn = (kt + 1) * 32;
      a0 = *(const float4*)(xp + kn);     a1 = *(const float4*)(xp + kn + 4);
      a2 = *(const float4*)(xp + kn + 8); a3 = *(const float4*)(xp + kn + 12);
      g0 = *(const float4*)(gp + kn);     g1 = *(const float4*)(gp + kn + 4);
      u0 = *(const float4*)(up + kn);     u1 = *(const float4*)(up + kn + 4);
    }
    __syncthreads();  // lds[cur] staged; lds[cur^1] consumed

    v16bf af0 = frag_ld(&lA[cur][(wr * 32) * LDST], lane);
    v16bf af1 = frag_ld(&lA[cur][(wr * 32 + 16) * LDST], lane);
    v16bf bg0 = frag_ld(&lG[cur][(wcol * 32) * LDST], lane);
    v16bf bg1 = frag_ld(&lG[cur][(wcol * 32 + 16) * LDST], lane);
    v16bf bu0 = frag_ld(&lU[cur][(wcol * 32) * LDST], lane);
    v16bf bu1 = frag_ld(&lU[cur][(wcol * 32 + 16) * LDST], lane);
    aG[0][0] = wmma_bf16(af0, bg0, aG[0][0]);
    aG[0][1] = wmma_bf16(af0, bg1, aG[0][1]);
    aG[1][0] = wmma_bf16(af1, bg0, aG[1][0]);
    aG[1][1] = wmma_bf16(af1, bg1, aG[1][1]);
    aU[0][0] = wmma_bf16(af0, bu0, aU[0][0]);
    aU[0][1] = wmma_bf16(af0, bu1, aU[0][1]);
    aU[1][0] = wmma_bf16(af1, bu0, aU[1][0]);
    aU[1][1] = wmma_bf16(af1, bu1, aU[1][1]);

    if (more) {
      const int nb = cur ^ 1;
      *(v8bf*)&lA[nb][ra * LDST + ca]     = cvt8(a0, a1);
      *(v8bf*)&lA[nb][ra * LDST + ca + 8] = cvt8(a2, a3);
      *(v8bf*)&lG[nb][rb * LDST + cb]     = cvt8(g0, g1);
      *(v8bf*)&lU[nb][rb * LDST + cb]     = cvt8(u0, u1);
    }
  }

  // epilogue: act = silu(g)*u -> LDS staging -> coalesced 16B global stores
  const int half = lane >> 4;
  const int nIdx = lane & 15;
#pragma unroll
  for (int mi = 0; mi < 2; ++mi) {
#pragma unroll
    for (int ni = 0; ni < 2; ++ni) {
      v8f g = aG[mi][ni], u = aU[mi][ni];
      const int col = wcol * 32 + ni * 16 + nIdx;
#pragma unroll
      for (int r = 0; r < 8; ++r) {
        const int mLoc = wr * 32 + mi * 16 + half * 8 + r;
        const float gv = g[r], uv = u[r];
        const float sv = gv / (1.f + __expf(-gv));
        sEpi[mLoc * EPST + col] = (__bf16)(sv * uv);
      }
    }
  }
  __syncthreads();
  {
    const int row = tid >> 1;
    const int cg = (tid & 1) * 32;
    if (mBase + row < c) {
      __bf16* dst = act + (long)sRow[row] * IDIM + i0 + cg;
#pragma unroll
      for (int j = 0; j < 4; ++j)
        *(uint4*)(dst + j * 8) = *(const uint4*)&sEpi[row * EPST + cg + j * 8];
    }
  }
}

// ---------------------------------------------------------------- down proj
// Block tile 128(M) x 128(H). Wave (wr,wcol): rows 32*wr..+31, cols 64*wcol..+63:
// 8 WMMA per K-step vs 12 ds_load_b128. A (bf16 act) is staged HBM->LDS with
// async loads (ASYNCcnt), B (fp32 weights) converts through VGPRs.
__global__ __launch_bounds__(256)
void moe_down_k(const __bf16* __restrict__ act, const float* __restrict__ down_w,
                const int* __restrict__ cnt, const int* __restrict__ pairTok,
                const float* __restrict__ pairW, float* __restrict__ out) {
  const int e = blockIdx.z;
  const int c = cnt[e];
  const int mBase = blockIdx.y * 128;
  if (mBase >= c) return;
  const int h0 = blockIdx.x * 128;

  __shared__ __bf16 lA[2][128 * LDST];
  __shared__ __bf16 lB[2][128 * LDST];
  __shared__ int sTok[128];
  __shared__ float sW[128];

  const int tid = threadIdx.x;
  if (tid < 128) {
    int idx = mBase + tid;
    if (idx > c - 1) idx = c - 1;
    sTok[tid] = pairTok[e * NTOK + idx];
    sW[tid] = pairW[e * NTOK + idx];
  }
  __syncthreads();

  const int ra = tid >> 1;                 // 0..127
  const int ca = (tid & 1) * 16;
  const __bf16* ap = act + (long)sTok[ra] * IDIM + ca;            // bf16 source
  const float* dp = down_w + (long)e * HDIM * IDIM + (long)(h0 + ra) * IDIM + ca;

  const unsigned ldsA0a = (unsigned)(size_t)&lA[0][ra * LDST + ca];
  const unsigned ldsA0b = ldsA0a + 16;
  const unsigned ldsA1a = (unsigned)(size_t)&lA[1][ra * LDST + ca];
  const unsigned ldsA1b = ldsA1a + 16;

  const int lane = tid & 31;
  const int wave = tid >> 5;
  const int wr = wave & 3;
  const int wcol = wave >> 2;

  v8f acc[2][4] = {{{}, {}, {}, {}}, {{}, {}, {}, {}}};

  // stage K-step 0
  async_ld_b128(ldsA0a, ap);
  async_ld_b128(ldsA0b, ap + 8);
  {
    float4 d0 = *(const float4*)(dp),     d1 = *(const float4*)(dp + 4);
    float4 d2 = *(const float4*)(dp + 8), d3 = *(const float4*)(dp + 12);
    *(v8bf*)&lB[0][ra * LDST + ca]     = cvt8(d0, d1);
    *(v8bf*)&lB[0][ra * LDST + ca + 8] = cvt8(d2, d3);
  }

  const int KS = IDIM / 32;
  for (int kt = 0; kt < KS; ++kt) {
    const int cur = kt & 1;
    float4 d0, d1, d2, d3;
    const bool more = (kt + 1) < KS;
    if (more) {
      const int kn = (kt + 1) * 32;
      d0 = *(const float4*)(dp + kn);     d1 = *(const float4*)(dp + kn + 4);
      d2 = *(const float4*)(dp + kn + 8); d3 = *(const float4*)(dp + kn + 12);
    }
    wait_asynccnt0();   // async LDS writes for lds[cur] landed
    __syncthreads();

    v16bf af0 = frag_ld(&lA[cur][(wr * 32) * LDST], lane);
    v16bf af1 = frag_ld(&lA[cur][(wr * 32 + 16) * LDST], lane);
#pragma unroll
    for (int ni = 0; ni < 4; ++ni) {
      v16bf bf = frag_ld(&lB[cur][(wcol * 64 + ni * 16) * LDST], lane);
      acc[0][ni] = wmma_bf16(af0, bf, acc[0][ni]);
      acc[1][ni] = wmma_bf16(af1, bf, acc[1][ni]);
    }

    if (more) {
      const int kn = (kt + 1) * 32;
      const int nb = cur ^ 1;
      async_ld_b128(nb ? ldsA1a : ldsA0a, ap + kn);
      async_ld_b128(nb ? ldsA1b : ldsA0b, ap + kn + 8);
      *(v8bf*)&lB[nb][ra * LDST + ca]     = cvt8(d0, d1);
      *(v8bf*)&lB[nb][ra * LDST + ca + 8] = cvt8(d2, d3);
    }
  }

  // epilogue: out[tok] += w * y (exactly 2 commutative fadds per element)
  const int half = lane >> 4;
  const int nIdx = lane & 15;
#pragma unroll
  for (int mi = 0; mi < 2; ++mi) {
#pragma unroll
    for (int ni = 0; ni < 4; ++ni) {
      v8f a = acc[mi][ni];
      const long col = h0 + wcol * 64 + ni * 16 + nIdx;
#pragma unroll
      for (int r = 0; r < 8; ++r) {
        const int mLoc = wr * 32 + mi * 16 + half * 8 + r;
        if (mBase + mLoc < c) {
          const long t = (long)(sTok[mLoc] >> 1);
          const float v = sW[mLoc] * a[r];
          __hip_atomic_fetch_add(&out[t * HDIM + col], v, __ATOMIC_RELAXED,
                                 __HIP_MEMORY_SCOPE_AGENT);
        }
      }
    }
  }
}

// ---------------------------------------------------------------- launcher
extern "C" void kernel_launch(void* const* d_in, const int* in_sizes, int n_in,
                              void* d_out, int out_size, void* d_ws,
                              size_t ws_size, hipStream_t stream) {
  (void)in_sizes; (void)n_in; (void)out_size; (void)ws_size;
  const float* x  = (const float*)d_in[0];   // [N,H]
  const float* rw = (const float*)d_in[1];   // [E,H]
  const float* gw = (const float*)d_in[2];   // [E,I,H]
  const float* uw = (const float*)d_in[3];   // [E,I,H]
  const float* dw = (const float*)d_in[4];   // [E,H,I]
  float* out = (float*)d_out;                // [N,H]

  char* ws = (char*)d_ws;
  int*    cnt     = (int*)ws;                                      // E ints
  int*    pairTok = (int*)(ws + 1024);                             // E*N ints
  float*  pairW   = (float*)(ws + 1024 + (size_t)NEXP * NTOK * 4); // E*N floats
  __bf16* act     = (__bf16*)(ws + 1024 + 2ull * NEXP * NTOK * 4); // N*2*I bf16

  zero_out_k<<<2048, 256, 0, stream>>>((float4*)out, NTOK * HDIM / 4, cnt);
  router_k<<<NTOK / 8, 256, 0, stream>>>(x, rw, cnt, pairTok, pairW);
  moe_gateup_k<<<dim3(IDIM / 64, NTOK / 128, NEXP), 256, 0, stream>>>(
      x, gw, uw, cnt, pairTok, act);
  moe_down_k<<<dim3(HDIM / 128, NTOK / 128, NEXP), 256, 0, stream>>>(
      act, dw, cnt, pairTok, pairW, out);
}
